// HardgroupAttention_16441134809373
// MI455X (gfx1250) — matrix operand order repaired and checked by reading, not compiled, not run on hardware
//
#include <hip/hip_runtime.h>
#include <hip/hip_bf16.h>

#define NHc 12
#define HDc 32
#define GPN 48
#define TOPKc 96
#define Nc 1024
#define Cc 384

typedef __attribute__((ext_vector_type(16))) __bf16 v16bf;
typedef __attribute__((ext_vector_type(8)))  float  v8f;

// ---------- WMMA tile load helpers (wave32, 16x16x32 bf16) ----------
// A-matrix 16x32 (MxK), contiguous k order used here: lane&15 = M row, lane>>4 picks K half.
// Lane holds K = {h*16..h*16+15} (16 contiguous).
__device__ __forceinline__ v16bf load_a_bf16(const __bf16* base, int stride) {
  const int l = threadIdx.x & 31;
  const int m = l & 15, hf = l >> 4;
  const __bf16* p = base + m * stride + hf * 16;
  v16bf r;
#pragma unroll
  for (int i = 0; i < 8; ++i) r[i] = p[i];
#pragma unroll
  for (int i = 0; i < 8; ++i) r[8 + i] = p[8 + i];
  return r;
}
__device__ __forceinline__ v16bf load_a_f32(const float* base, int stride) {
  const int l = threadIdx.x & 31;
  const int m = l & 15, hf = l >> 4;
  const float* p = base + m * stride + hf * 16;
  v16bf r;
#pragma unroll
  for (int i = 0; i < 8; ++i) r[i] = (__bf16)p[i];
#pragma unroll
  for (int i = 0; i < 8; ++i) r[8 + i] = (__bf16)p[8 + i];
  return r;
}
// B-matrix 32x16 (KxN) loaded from Bt (NxK row-major): lane&15 = N column,
// element e holds K = (lane>>4)*16 + e (16 contiguous).
__device__ __forceinline__ v16bf load_b_bf16(const __bf16* base, int stride) {
  const int l = threadIdx.x & 31;
  const int n = l & 15, hf = l >> 4;
  const __bf16* p = base + n * stride + hf * 16;
  v16bf r;
#pragma unroll
  for (int i = 0; i < 16; ++i) r[i] = p[i];
  return r;
}
__device__ __forceinline__ v16bf load_b_f32(const float* base, int stride) {
  const int l = threadIdx.x & 31;
  const int n = l & 15, hf = l >> 4;
  const float* p = base + n * stride + hf * 16;
  v16bf r;
#pragma unroll
  for (int i = 0; i < 16; ++i) r[i] = (__bf16)p[i];
  return r;
}
__device__ __forceinline__ v8f wmma_bf16(v16bf a, v16bf b, v8f c) {
  return __builtin_amdgcn_wmma_f32_16x16x32_bf16(false, a, false, b, (short)0, c,
                                                 false, false);
}

// ---------- K0: qkv = x @ qkv_w^T, scatter into q/k/v^T bf16 buffers ----------
__global__ void k_qkv(const float* __restrict__ x, const float* __restrict__ w,
                      __bf16* __restrict__ qb, __bf16* __restrict__ kb,
                      __bf16* __restrict__ vt) {
  const int wid = blockIdx.x * (blockDim.x >> 5) + (threadIdx.x >> 5);
  const int mt = wid / 72, ct = wid % 72;        // 256 x 72 tiles
  const int row0 = mt * 16, col0 = ct * 16;
  v8f acc = {};
  for (int kk = 0; kk < Cc; kk += 32) {
    v16bf a = load_a_f32(x + row0 * Cc + kk, Cc);
    v16bf b = load_b_f32(w + col0 * Cc + kk, Cc);
    acc = wmma_bf16(a, b, acc);
  }
  const int l = threadIdx.x & 31, hf = l >> 4, cn = l & 15;
  const int col = col0 + cn;
  const int sec = col / Cc;           // 0=q 1=k 2=v (uniform per tile)
  const int hh  = (col % Cc) / HDc;   // head (uniform per tile)
  const int d   = col % HDc;
#pragma unroll
  for (int r = 0; r < 8; ++r) {
    const int gr = row0 + r + 8 * hf;
    const int b_ = gr >> 10, n = gr & 1023;
    const __bf16 v = (__bf16)acc[r];
    const int bh = b_ * NHc + hh;
    if (sec == 0)      qb[(bh * Nc + n) * HDc + d] = v;
    else if (sec == 1) kb[(bh * Nc + n) * HDc + d] = v;
    else               vt[(bh * HDc + d) * Nc + n] = v;     // transposed for attn@v
  }
}

// ---------- K1: top-1 group routing + group accumulation ----------
__global__ void k_route(const __bf16* __restrict__ qb,
                        const float* __restrict__ gp_w,
                        float* __restrict__ qsum, float* __restrict__ gcnt) {
  __shared__ float gp[GPN * HDc];
  const int bh = blockIdx.x;           // 0..47
  const int h = bh % NHc;
  for (int i = threadIdx.x; i < GPN * HDc; i += blockDim.x)
    gp[i] = gp_w[h * GPN * HDc + i];   // gp[(h*48+g)*32+d]
  __syncthreads();
  const int n = threadIdx.x;           // 1024 threads = 1024 queries
  float q[HDc];
#pragma unroll
  for (int d = 0; d < HDc; ++d) q[d] = (float)qb[(bh * Nc + n) * HDc + d];
  float best = -3.4e38f; int bg = 0;
  for (int g = 0; g < GPN; ++g) {
    float s = 0.f;
#pragma unroll
    for (int d = 0; d < HDc; ++d) s += q[d] * gp[g * HDc + d];
    if (s > best) { best = s; bg = g; }
  }
  float* qs = qsum + (bh * GPN + bg) * HDc;
  for (int d = 0; d < HDc; ++d) atomicAdd(&qs[d], q[d]);
  atomicAdd(&gcnt[bh * GPN + bg], 1.0f);
}

// ---------- K2: q_mean·k^T per group, top-96 keys -> col counts ----------
__global__ void k_topk(const __bf16* __restrict__ kb,
                       const float* __restrict__ qsum,
                       const float* __restrict__ gcnt,
                       float* __restrict__ colcnt) {
  __shared__ float qm[HDc];
  __shared__ float tv[Nc];
  __shared__ float rv[256];
  __shared__ int   ri[256];
  const int gid = blockIdx.x;               // bh*48 + g
  const int bh = gid / GPN, g = gid % GPN;
  const int t = threadIdx.x;
  if (t < HDc) {
    const float c = gcnt[bh * GPN + g];
    qm[t] = (c > 0.f) ? qsum[(bh * GPN + g) * HDc + t] / c : 0.f;
  }
  __syncthreads();
  for (int m = t; m < Nc; m += 256) {
    float s = 0.f;
    const __bf16* kr = kb + (bh * Nc + m) * HDc;
#pragma unroll
    for (int d = 0; d < HDc; ++d) s += qm[d] * (float)kr[d];
    tv[m] = s;
  }
  __syncthreads();
  for (int it = 0; it < TOPKc; ++it) {
    float best = -3.4e38f; int bi = 0;
    for (int m = t; m < Nc; m += 256) {
      const float v = tv[m];
      if (v > best) { best = v; bi = m; }
    }
    rv[t] = best; ri[t] = bi;
    __syncthreads();
    for (int s = 128; s > 0; s >>= 1) {
      if (t < s) {
        if (rv[t + s] > rv[t] || (rv[t + s] == rv[t] && ri[t + s] < ri[t])) {
          rv[t] = rv[t + s]; ri[t] = ri[t + s];
        }
      }
      __syncthreads();
    }
    if (t == 0) {
      tv[ri[0]] = -3.4e38f;
      atomicAdd(&colcnt[bh * Nc + ri[0]], 1.0f);
    }
    __syncthreads();
  }
}

// ---------- K3: online softmax row stats (rowmax, rowsumexp) via WMMA ----------
__global__ void k_rowstats(const __bf16* __restrict__ qb,
                           const __bf16* __restrict__ kb,
                           float* __restrict__ rowmax, float* __restrict__ rowsum) {
  const int wid = blockIdx.x * (blockDim.x >> 5) + (threadIdx.x >> 5);
  const int bh = wid >> 6, nt = wid & 63;
  const int n0 = nt * 16;
  const float scale = 0.17677669529663687f;
  const v16bf a = load_a_bf16(qb + (bh * Nc + n0) * HDc, HDc);
  float rm[8], rs[8];
#pragma unroll
  for (int r = 0; r < 8; ++r) { rm[r] = -3.4e38f; rs[r] = 0.f; }
  for (int mt = 0; mt < 64; ++mt) {
    const v16bf bm = load_b_bf16(kb + (bh * Nc + mt * 16) * HDc, HDc);
    v8f c = {};
    c = wmma_bf16(a, bm, c);
#pragma unroll
    for (int r = 0; r < 8; ++r) {
      const float v = c[r] * scale;
      float tmax = v;
      tmax = fmaxf(tmax, __shfl_xor(tmax, 1, 32));
      tmax = fmaxf(tmax, __shfl_xor(tmax, 2, 32));
      tmax = fmaxf(tmax, __shfl_xor(tmax, 4, 32));
      tmax = fmaxf(tmax, __shfl_xor(tmax, 8, 32));
      const float nm = fmaxf(rm[r], tmax);
      float e = __expf(v - nm);
      e += __shfl_xor(e, 1, 32);
      e += __shfl_xor(e, 2, 32);
      e += __shfl_xor(e, 4, 32);
      e += __shfl_xor(e, 8, 32);
      rs[r] = rs[r] * __expf(rm[r] - nm) + e;
      rm[r] = nm;
    }
  }
  const int l = threadIdx.x & 31, hf = l >> 4;
#pragma unroll
  for (int r = 0; r < 8; ++r) {
    if ((l & 15) == r) {
      const int n = n0 + r + 8 * hf;
      rowmax[bh * Nc + n] = rm[r];
      rowsum[bh * Nc + n] = rs[r];
    }
  }
}

// ---------- K4: S[m] = column sums of row-softmax P ----------
__global__ void k_colsum(const __bf16* __restrict__ qb,
                         const __bf16* __restrict__ kb,
                         const float* __restrict__ rowmax,
                         const float* __restrict__ rowsum,
                         float* __restrict__ Scol) {
  const int wid = blockIdx.x * (blockDim.x >> 5) + (threadIdx.x >> 5);
  const int bh = wid >> 6, nt = wid & 63;
  const int n0 = nt * 16;
  const float scale = 0.17677669529663687f;
  const int l = threadIdx.x & 31, hf = l >> 4;
  const v16bf a = load_a_bf16(qb + (bh * Nc + n0) * HDc, HDc);
  float rm[8], irs[8];
#pragma unroll
  for (int r = 0; r < 8; ++r) {
    const int n = n0 + r + 8 * hf;
    rm[r] = rowmax[bh * Nc + n];
    irs[r] = 1.0f / rowsum[bh * Nc + n];
  }
  for (int mt = 0; mt < 64; ++mt) {
    const v16bf bm = load_b_bf16(kb + (bh * Nc + mt * 16) * HDc, HDc);
    v8f c = {};
    c = wmma_bf16(a, bm, c);
    float part = 0.f;
#pragma unroll
    for (int r = 0; r < 8; ++r)
      part += __expf(c[r] * scale - rm[r]) * irs[r];
    part += __shfl_xor(part, 16, 32);
    if (l < 16) atomicAdd(&Scol[bh * Nc + mt * 16 + l], part);
  }
}

// ---------- K6: out = (P * w[m]) @ v via WMMA, LDS-staged P tiles ----------
__global__ void k_attn_out(const __bf16* __restrict__ qb,
                           const __bf16* __restrict__ kb,
                           const __bf16* __restrict__ vt,
                           const float* __restrict__ rowmax,
                           const float* __restrict__ rowsum,
                           const float* __restrict__ colcnt,
                           const float* __restrict__ Scol,
                           __bf16* __restrict__ ao) {
  __shared__ __bf16 lp[4][16 * 32];
  const int wv = threadIdx.x >> 5;
  const int wid = blockIdx.x * 4 + wv;
  const int bh = wid >> 6, nt = wid & 63;
  const int b = bh / NHc, h = bh % NHc;
  const int n0 = nt * 16;
  const int l = threadIdx.x & 31, hf = l >> 4, cn = l & 15;
  const float scale = 0.17677669529663687f;
  const v16bf aq = load_a_bf16(qb + (bh * Nc + n0) * HDc, HDc);
  float rm[8], irs[8];
#pragma unroll
  for (int r = 0; r < 8; ++r) {
    const int n = n0 + r + 8 * hf;
    rm[r] = rowmax[bh * Nc + n];
    irs[r] = 1.0f / rowsum[bh * Nc + n];
  }
  v8f acc0 = {}, acc1 = {};
  __bf16* myp = lp[wv];
  for (int mc = 0; mc < 32; ++mc) {
    const int mbase = mc * 32;
#pragma unroll
    for (int sub = 0; sub < 2; ++sub) {
      const int m0 = mbase + sub * 16;
      const v16bf bk = load_b_bf16(kb + (bh * Nc + m0) * HDc, HDc);
      v8f c = {};
      c = wmma_bf16(aq, bk, c);
      const float cc = colcnt[bh * Nc + m0 + cn];
      const float ss = Scol[bh * Nc + m0 + cn];
      const float w = cc / (cc * ss + 1e-8f);
#pragma unroll
      for (int r = 0; r < 8; ++r) {
        const float p = __expf(c[r] * scale - rm[r]) * irs[r] * w;
        myp[(r + 8 * hf) * 32 + sub * 16 + cn] = (__bf16)p;
      }
    }
    asm volatile("" ::: "memory");   // keep LDS store->load order; DS is in-order per wave
    v16bf ap;
    {
      const __bf16* p = myp + cn * 32 + hf * 16;   // A-layout read of P tile
#pragma unroll
      for (int i = 0; i < 8; ++i) ap[i] = p[i];
#pragma unroll
      for (int i = 0; i < 8; ++i) ap[8 + i] = p[8 + i];
    }
    const v16bf bv0 = load_b_bf16(vt + (bh * HDc + 0)  * Nc + mbase, Nc);
    const v16bf bv1 = load_b_bf16(vt + (bh * HDc + 16) * Nc + mbase, Nc);
    acc0 = wmma_bf16(ap, bv0, acc0);
    acc1 = wmma_bf16(ap, bv1, acc1);
  }
#pragma unroll
  for (int r = 0; r < 8; ++r) {
    const int n = n0 + r + 8 * hf;
    __bf16* o = ao + ((b * Nc + n) * NHc + h) * HDc;   // [b,n,h,d] for final proj
    o[cn] = (__bf16)acc0[r];
    o[16 + cn] = (__bf16)acc1[r];
  }
}

// ---------- K7: final projection out = ao @ proj_w^T ----------
__global__ void k_proj(const __bf16* __restrict__ ao,
                       const float* __restrict__ pw, float* __restrict__ out) {
  const int wid = blockIdx.x * (blockDim.x >> 5) + (threadIdx.x >> 5);
  const int mt = wid / 24, ct = wid % 24;   // 256 x 24 tiles
  const int row0 = mt * 16, col0 = ct * 16;
  v8f acc = {};
  for (int kk = 0; kk < Cc; kk += 32) {
    v16bf a = load_a_bf16(ao + row0 * Cc + kk, Cc);
    v16bf b = load_b_f32(pw + col0 * Cc + kk, Cc);
    acc = wmma_bf16(a, b, acc);
  }
  const int l = threadIdx.x & 31, hf = l >> 4, cn = l & 15;
#pragma unroll
  for (int r = 0; r < 8; ++r)
    out[(row0 + r + 8 * hf) * Cc + col0 + cn] = acc[r];
}

extern "C" void kernel_launch(void* const* d_in, const int* in_sizes, int n_in,
                              void* d_out, int out_size, void* d_ws, size_t ws_size,
                              hipStream_t stream) {
  const float* x      = (const float*)d_in[0];
  const float* qkv_w  = (const float*)d_in[1];
  const float* gp_w   = (const float*)d_in[2];
  const float* proj_w = (const float*)d_in[3];
  float* out = (float*)d_out;
  char* ws = (char*)d_ws;

  // bf16 tensors: 4*12*1024*32*2 = 3,145,728 B each
  __bf16* qb = (__bf16*)(ws + 0);
  __bf16* kb = (__bf16*)(ws + 3145728);
  __bf16* vt = (__bf16*)(ws + 6291456);
  __bf16* ao = (__bf16*)(ws + 9437184);
  // fp32 scratch (zero-initialized region)
  float* qsum   = (float*)(ws + 12582912);  // 4*12*48*32*4 = 294,912
  float* gcnt   = (float*)(ws + 12877824);  // 4*12*48*4   =   9,216
  float* colcnt = (float*)(ws + 12887040);  // 4*12*1024*4 = 196,608
  float* Scol   = (float*)(ws + 13083648);  // 196,608
  float* rowmax = (float*)(ws + 13280256);  // written before read
  float* rowsum = (float*)(ws + 13476864);

  hipMemsetAsync(ws + 12582912, 0, 697344, stream);

  k_qkv     <<<4608, 128, 0, stream>>>(x, qkv_w, qb, kb, vt);
  k_route   <<<48,  1024, 0, stream>>>(qb, gp_w, qsum, gcnt);
  k_topk    <<<2304, 256, 0, stream>>>(kb, qsum, gcnt, colcnt);
  k_rowstats<<<768,  128, 0, stream>>>(qb, kb, rowmax, rowsum);
  k_colsum  <<<768,  128, 0, stream>>>(qb, kb, rowmax, rowsum, Scol);
  k_attn_out<<<768,  128, 0, stream>>>(qb, kb, vt, rowmax, rowsum, colcnt, Scol, ao);
  k_proj    <<<1536, 128, 0, stream>>>(ao, proj_w, out);
}
